// MultiHeadAttention_90529320665164
// MI455X (gfx1250) — matrix operand order repaired and checked by reading, not compiled, not run on hardware
//
#include <hip/hip_runtime.h>
#include <hip/hip_bf16.h>
#include <stdint.h>

#define EMBED  1024
#define NHEAD  16
#define HDIM   64
#define BATCH  4
#define SEQ    2048
#define TOKENS (BATCH * SEQ) // 8192

typedef __bf16 bf16_t;
typedef bf16_t v16bf __attribute__((ext_vector_type(16)));
typedef float  v8f   __attribute__((ext_vector_type(8)));

union Frag { v16bf v; unsigned u[8]; };
union CF   { v8f   v; float    f[8]; };

// fp32 -> bf16 round-to-nearest-even
__device__ __forceinline__ unsigned short f2bf(float x) {
  unsigned u = __builtin_bit_cast(unsigned, x);
  unsigned r = u + 0x7FFFu + ((u >> 16) & 1u);
  return (unsigned short)(r >> 16);
}

// 32-bit load of a bf16 pair (index must be even)
__device__ __forceinline__ unsigned ldu(const unsigned short* p, size_t i) {
  return *(const unsigned*)(p + i);
}

// A-fragment dword v holds K = 2v + (v>=4 ? 8 : 0) + 8*hi  (16-bit A 16x32 layout)
__device__ __forceinline__ int kA(int v, int hi) { return 2*v + ((v >= 4) ? 8 : 0) + 8*hi; }
// B-fragment dword v holds K = 2v + (v>=4 ? 8 : 0) + 8*hi  (same K order as A)
__device__ __forceinline__ int kB(int v, int hi) { return 2*v + ((v >= 4) ? 8 : 0) + 8*hi; }

__device__ __forceinline__ v8f wmma_bf16(v16bf a, v16bf b, v8f c) {
  return __builtin_amdgcn_wmma_f32_16x16x32_bf16(false, a, false, b, (short)0, c,
                                                 false, false);
}

// ---------------------------------------------------------------- converts
__global__ void cvt_f32_bf16(const float* __restrict__ in,
                             unsigned short* __restrict__ out, int n) {
  for (int i = blockIdx.x * blockDim.x + threadIdx.x; i < n;
       i += gridDim.x * blockDim.x)
    out[i] = f2bf(in[i]);
}

// in is K x N (row-major), out is N x K (row-major) bf16
__global__ void cvt_transpose_bf16(const float* __restrict__ in,
                                   unsigned short* __restrict__ out,
                                   int K, int N) {
  int total = K * N;
  for (int i = blockIdx.x * blockDim.x + threadIdx.x; i < total;
       i += gridDim.x * blockDim.x) {
    int n = i / K, k = i % K;
    out[i] = f2bf(in[(size_t)k * N + n]);
  }
}

// ------------------------------------------------------- shared GEMM mainloop
// Per wave: 32(M) x 64(N) C tile, K = EMBED, software-pipelined (double buffer).
// A: [M][K] bf16 row-major (K contiguous).  B: [N][K] bf16 (pre-transposed).
__device__ __forceinline__ void gemm_mainloop_32x64(
    const unsigned short* __restrict__ Apt,
    const unsigned short* __restrict__ Bpt,
    const size_t arow[2], const size_t brow[4], int hi, CF C[2][4]) {
  Frag a0[2], b0[4], a1[2], b1[4];

#define LOAD_A(dst, kc)                                                       \
  {                                                                           \
    _Pragma("unroll") for (int i = 0; i < 2; i++)                             \
        _Pragma("unroll") for (int v = 0; v < 8; v++)                         \
            dst[i].u[v] = ldu(Apt, arow[i] + (kc) + kA(v, hi));               \
  }
#define LOAD_B(dst, kc)                                                       \
  {                                                                           \
    _Pragma("unroll") for (int j = 0; j < 4; j++)                             \
        _Pragma("unroll") for (int v = 0; v < 8; v++)                         \
            dst[j].u[v] = ldu(Bpt, brow[j] + (kc) + kB(v, hi));               \
  }
#define MMA(a, b)                                                             \
  {                                                                           \
    _Pragma("unroll") for (int j = 0; j < 4; j++)                             \
        _Pragma("unroll") for (int i = 0; i < 2; i++)                         \
            C[i][j].v = wmma_bf16(a[i].v, b[j].v, C[i][j].v);                 \
  }

  LOAD_A(a0, 0);
  LOAD_B(b0, 0);
  for (int kc = 0; kc + 64 <= EMBED; kc += 64) {
    // issue next chunk's loads before consuming the current one
    LOAD_A(a1, kc + 32);
    LOAD_B(b1, kc + 32);
    MMA(a0, b0);
    if (kc + 64 < EMBED) {
      LOAD_A(a0, kc + 64);
      LOAD_B(b0, kc + 64);
    }
    MMA(a1, b1);
  }
#undef LOAD_A
#undef LOAD_B
#undef MMA
}

// ---------------------------------------------------------------- QKV GEMM
// xb: [8192][1024] bf16, wT: [3072][1024] bf16 (transposed), bias f32[3072]
// Writes Qb/Kb as [b*H+h][s][64] bf16, Vt as [b*H+h][64][2048] bf16.
__global__ void __launch_bounds__(256) qkv_gemm_kernel(
    const unsigned short* __restrict__ xb,
    const unsigned short* __restrict__ wT,
    const float* __restrict__ bias,
    unsigned short* __restrict__ Qb,
    unsigned short* __restrict__ Kb,
    unsigned short* __restrict__ Vt) {
  const int wid = threadIdx.x >> 5;
  const int lane = threadIdx.x & 31;
  const int hi = lane >> 4;
  const int ln = lane & 15;
  const int gw = blockIdx.x * 8 + wid;   // 12288 waves
  const int mT = gw / 48, nT = gw % 48;  // 256 x 48 tiles
  const int m0 = mT * 32, n0 = nT * 64;

  CF C[2][4];
  for (int i = 0; i < 2; i++)
    for (int j = 0; j < 4; j++)
      for (int r = 0; r < 8; r++) C[i][j].f[r] = 0.f;

  size_t arow[2], brow[4];
  for (int i = 0; i < 2; i++) arow[i] = (size_t)(m0 + 16 * i + ln) * EMBED;
  for (int j = 0; j < 4; j++) brow[j] = (size_t)(n0 + 16 * j + ln) * EMBED;

  gemm_mainloop_32x64(xb, wT, arow, brow, hi, C);

  const int sel = n0 >> 10;      // 0=Q 1=K 2=V
  const int h = (n0 >> 6) & 15;  // head
#pragma unroll
  for (int j = 0; j < 4; j++) {
    const int d = 16 * j + ln;
    const float bval = bias[n0 + 16 * j + ln];
#pragma unroll
    for (int i = 0; i < 2; i++) {
#pragma unroll
      for (int r = 0; r < 8; r++) {
        const int tok = m0 + 16 * i + r + 8 * hi;
        const int bb = tok >> 11;
        const int s = tok & (SEQ - 1);
        const unsigned short val = f2bf(C[i][j].f[r] + bval);
        const size_t hb = (size_t)(bb * NHEAD + h) * SEQ * HDIM;
        if (sel == 0)      Qb[hb + (size_t)s * HDIM + d] = val;
        else if (sel == 1) Kb[hb + (size_t)s * HDIM + d] = val;
        else               Vt[hb + (size_t)d * SEQ + s] = val;
      }
    }
  }
}

// ---------------------------------------------------------------- attention
// Per wave: 16 queries of one (b,h), flash-style online softmax over K tiles.
__global__ void __launch_bounds__(256) attn_kernel(
    const unsigned short* __restrict__ Qb,
    const unsigned short* __restrict__ Kb,
    const unsigned short* __restrict__ Vt,
    unsigned short* __restrict__ Obuf) {
  __shared__ unsigned short Plds[8][16][32];  // per-wave 16x32 P tile
  const int wid = threadIdx.x >> 5;
  const int lane = threadIdx.x & 31;
  const int hi = lane >> 4;
  const int ln = lane & 15;
  const int bh = blockIdx.x >> 4;   // 0..63
  const int qblk = blockIdx.x & 15; // 0..15
  const int q0 = qblk * 128 + wid * 16;
  const int b = bh >> 4, h = bh & 15;

  const unsigned short* Qh = Qb + (size_t)bh * SEQ * HDIM;
  const unsigned short* Kh = Kb + (size_t)bh * SEQ * HDIM;
  const unsigned short* Vh = Vt + (size_t)bh * SEQ * HDIM; // [d][s]

  Frag aq[2]; // Q tile: 16 x 64, two K=32 chunks, resident all pass
  {
    const size_t rbase = (size_t)(q0 + ln) * HDIM;
#pragma unroll
    for (int c = 0; c < 2; c++)
#pragma unroll
      for (int v = 0; v < 8; v++)
        aq[c].u[v] = ldu(Qh, rbase + 32 * c + kA(v, hi));
  }

  float m[8], l[8], alpha[8];
  CF O[4];
  for (int r = 0; r < 8; r++) { m[r] = -INFINITY; l[r] = 0.f; }
  for (int n = 0; n < 4; n++)
    for (int r = 0; r < 8; r++) O[n].f[r] = 0.f;

  for (int k0 = 0; k0 < SEQ; k0 += 32) {
    // ---- scores: two 16x16 tiles covering 32 keys
    CF S[2];
    for (int t = 0; t < 2; t++)
      for (int r = 0; r < 8; r++) S[t].f[r] = 0.f;
#pragma unroll
    for (int t = 0; t < 2; t++) {
      const size_t tokbase = (size_t)(k0 + 16 * t + ln) * HDIM;
#pragma unroll
      for (int c = 0; c < 2; c++) {
        Frag bk;
#pragma unroll
        for (int v = 0; v < 8; v++)
          bk.u[v] = ldu(Kh, tokbase + 32 * c + kB(v, hi));
        S[t].v = wmma_bf16(aq[c].v, bk.v, S[t].v);
      }
    }
    // ---- hoist V fragment loads: latency hidden under the softmax VALU work
    Frag bv[4];
#pragma unroll
    for (int n = 0; n < 4; n++) {
      const size_t dbase = (size_t)(16 * n + ln) * SEQ + k0;
#pragma unroll
      for (int v = 0; v < 8; v++) bv[n].u[v] = ldu(Vh, dbase + kB(v, hi));
    }
    // ---- prefetch next iteration's K/V tiles (global_prefetch_b8)
    if (k0 + 32 < SEQ) {
      __builtin_prefetch(Kh + (size_t)(k0 + 32 + lane) * HDIM, 0, 3);
      __builtin_prefetch(Vh + (size_t)(2 * lane) * SEQ + k0 + 32, 0, 3);
      __builtin_prefetch(Vh + (size_t)(2 * lane + 1) * SEQ + k0 + 32, 0, 3);
    }
    // ---- online softmax: rows r+8*hi live in a 16-lane half -> xor masks 1..8
#pragma unroll
    for (int r = 0; r < 8; r++) {
      float s0 = S[0].f[r] * 0.125f;  // 1/sqrt(64)
      float s1 = S[1].f[r] * 0.125f;
      float mx = fmaxf(s0, s1);
      for (int msk = 8; msk >= 1; msk >>= 1)
        mx = fmaxf(mx, __shfl_xor(mx, msk, 32));
      float mnew = fmaxf(m[r], mx);
      float a = __expf(m[r] - mnew);
      float p0 = __expf(s0 - mnew);
      float p1 = __expf(s1 - mnew);
      float rs = p0 + p1;
      for (int msk = 8; msk >= 1; msk >>= 1)
        rs += __shfl_xor(rs, msk, 32);
      l[r] = l[r] * a + rs;
      m[r] = mnew;
      alpha[r] = a;
      S[0].f[r] = p0;
      S[1].f[r] = p1;
    }
    for (int n = 0; n < 4; n++)
      for (int r = 0; r < 8; r++) O[n].f[r] *= alpha[r];
    // ---- stage P (16x32 bf16) through LDS to build the A fragment
#pragma unroll
    for (int t = 0; t < 2; t++)
#pragma unroll
      for (int r = 0; r < 8; r++)
        Plds[wid][r + 8 * hi][ln + 16 * t] = f2bf(S[t].f[r]);
    __syncthreads();
    Frag ap;
#pragma unroll
    for (int v = 0; v < 8; v++)
      ap.u[v] = *(const unsigned*)&Plds[wid][ln][kA(v, hi)];
    // ---- O(16x64) += P(16x32) @ V(32x64); V fragments already resident
#pragma unroll
    for (int n = 0; n < 4; n++) O[n].v = wmma_bf16(ap.v, bv[n].v, O[n].v);
    __syncthreads();
  }

  float inv[8];
  for (int r = 0; r < 8; r++) inv[r] = 1.0f / l[r];
#pragma unroll
  for (int n = 0; n < 4; n++) {
    const int d = 16 * n + ln;
#pragma unroll
    for (int r = 0; r < 8; r++) {
      const int tok = b * SEQ + q0 + r + 8 * hi;
      Obuf[(size_t)tok * EMBED + h * HDIM + d] = f2bf(O[n].f[r] * inv[r]);
    }
  }
}

// ---------------------------------------------------------------- out GEMM
__global__ void __launch_bounds__(256) out_gemm_kernel(
    const unsigned short* __restrict__ Ob,
    const unsigned short* __restrict__ wT,
    const float* __restrict__ bias,
    float* __restrict__ out) {
  const int wid = threadIdx.x >> 5;
  const int lane = threadIdx.x & 31;
  const int hi = lane >> 4;
  const int ln = lane & 15;
  const int gw = blockIdx.x * 8 + wid;  // 4096 waves
  const int mT = gw >> 4, nT = gw & 15; // 256 x 16 tiles
  const int m0 = mT * 32, n0 = nT * 64;

  CF C[2][4];
  for (int i = 0; i < 2; i++)
    for (int j = 0; j < 4; j++)
      for (int r = 0; r < 8; r++) C[i][j].f[r] = 0.f;

  size_t arow[2], brow[4];
  for (int i = 0; i < 2; i++) arow[i] = (size_t)(m0 + 16 * i + ln) * EMBED;
  for (int j = 0; j < 4; j++) brow[j] = (size_t)(n0 + 16 * j + ln) * EMBED;

  gemm_mainloop_32x64(Ob, wT, arow, brow, hi, C);

#pragma unroll
  for (int j = 0; j < 4; j++) {
    const float bval = bias[n0 + 16 * j + ln];
#pragma unroll
    for (int i = 0; i < 2; i++) {
#pragma unroll
      for (int r = 0; r < 8; r++) {
        const int tok = m0 + 16 * i + r + 8 * hi;
        out[(size_t)tok * EMBED + n0 + 16 * j + ln] = C[i][j].f[r] + bval;
      }
    }
  }
}

// ---------------------------------------------------------------- launch
extern "C" void kernel_launch(void* const* d_in, const int* in_sizes, int n_in,
                              void* d_out, int out_size, void* d_ws,
                              size_t ws_size, hipStream_t stream) {
  const float* x     = (const float*)d_in[0];
  const float* w_qkv = (const float*)d_in[1];
  const float* b_qkv = (const float*)d_in[2];
  const float* w_out = (const float*)d_in[3];
  const float* b_out = (const float*)d_in[4];
  float* out = (float*)d_out;

  // workspace layout (bytes), total 92,274,688 (~88 MB)
  char* ws = (char*)d_ws;
  unsigned short* xb    = (unsigned short*)(ws + 0);         // 8192x1024 bf16
  unsigned short* wqkvT = (unsigned short*)(ws + 16777216);  // 3072x1024 bf16
  unsigned short* woutT = (unsigned short*)(ws + 23068672);  // 1024x1024 bf16
  unsigned short* Qb    = (unsigned short*)(ws + 25165824);  // [64][2048][64]
  unsigned short* Kb    = (unsigned short*)(ws + 41943040);  // [64][2048][64]
  unsigned short* Vt    = (unsigned short*)(ws + 58720256);  // [64][64][2048]
  unsigned short* Obuf  = (unsigned short*)(ws + 75497472);  // 8192x1024 bf16

  cvt_f32_bf16<<<2048, 256, 0, stream>>>(x, xb, TOKENS * EMBED);
  cvt_transpose_bf16<<<2048, 256, 0, stream>>>(w_qkv, wqkvT, EMBED, 3 * EMBED);
  cvt_transpose_bf16<<<1024, 256, 0, stream>>>(w_out, woutT, EMBED, EMBED);
  qkv_gemm_kernel<<<1536, 256, 0, stream>>>(xb, wqkvT, b_qkv, Qb, Kb, Vt);
  attn_kernel<<<1024, 256, 0, stream>>>(Qb, Kb, Vt, Obuf);
  out_gemm_kernel<<<512, 256, 0, stream>>>(Obuf, woutT, b_out, out);
}
